// STHGNN_87058987090148
// MI455X (gfx1250) — matrix operand order repaired and checked by reading, not compiled, not run on hardware
//
#include <hip/hip_runtime.h>

typedef unsigned short u16;
typedef unsigned int   u32;
typedef __attribute__((ext_vector_type(16))) __bf16 v16bf;
typedef __attribute__((ext_vector_type(8)))  float  v8f;
typedef __attribute__((ext_vector_type(8)))  unsigned int v8u;
typedef __attribute__((ext_vector_type(4)))  int v4i;
typedef __attribute__((address_space(1))) v4i gv4i;   // global int4 vector
typedef __attribute__((address_space(3))) v4i lv4i;   // LDS int4 vector

#define DEV __device__ __forceinline__

#ifndef __has_builtin
#define __has_builtin(x) 0
#endif
#if __has_builtin(__builtin_amdgcn_global_load_async_to_lds_b128)
#define ASYNC_LDS 1
#endif
#if __has_builtin(__builtin_amdgcn_s_wait_asynccnt)
#define WAIT_ASYNC() __builtin_amdgcn_s_wait_asynccnt(0)
#else
#define WAIT_ASYNC() asm volatile("s_wait_asynccnt 0x0" ::: "memory")
#endif

DEV u16 f2bf(float f) {                       // round-to-nearest-even fp32 -> bf16
  u32 u = __builtin_bit_cast(u32, f);
  u32 r = u + 0x7FFFu + ((u >> 16) & 1u);
  return (u16)(r >> 16);
}
DEV float sigm(float x) { return 1.0f / (1.0f + __expf(-x)); }

// ---------------------------------------------------------------------------
// Generic bf16 WMMA GEMM:  C[b] = acc?C[b]: 0  +  alpha * A[b] @ B[b]  + bias
//   A: bf16 row-major [R,K]  (per-batch offset = (batch/aDiv)*aBS)
//   B: bf16 row-major [K,N]  (per-batch offset = batch*bBS)
//   C: fp32 (optional) and bf16 (optional) row-major [R,N], offset batch*cBS
// R%64==0, K%64==0, N%64==0. 4 waves/block; block computes a 64x64 C tile.
// K slabs of 64 are double-buffered in LDS; the A slab is filled with
// GLOBAL_LOAD_ASYNC_TO_LDS_B128 (ASYNCcnt) when the toolchain exposes it.
// ---------------------------------------------------------------------------
#define GEMM_BIAS 1
#define GEMM_ACC  2
#define TP 72   // LDS pitch (bf16 elems) = 144B: 16B aligned, bank-conflict free

// stage 64x64 bf16 A slab (row-major) into LDS [r*TP + k]
DEV void stage_a(u16* dst, const u16* Ag, int K, int tid) {
#pragma unroll
  for (int i = 0; i < 4; ++i) {
    int c = tid + (i << 7);
    int r = c >> 3, cc = (c & 7) << 3;
#ifdef ASYNC_LDS
    __builtin_amdgcn_global_load_async_to_lds_b128(
        (gv4i*)(Ag + (long)r * K + cc),
        (lv4i*)(dst + r * TP + cc), 0, 0);
#else
    uint4 v = *(const uint4*)(Ag + (long)r * K + cc);
    *(uint4*)(dst + r * TP + cc) = v;
#endif
  }
}
// stage 64(k) x 64(n) bf16 B slab transposed into LDS [n*TP + k]
DEV void stage_b(u16* dst, const u16* Bg, int N, int tid) {
#pragma unroll
  for (int i = 0; i < 4; ++i) {
    int c = tid + (i << 7);
    int k = c >> 3, nc = (c & 7) << 3;
    uint4 v = *(const uint4*)(Bg + (long)k * N + nc);
    u16 e[8]; *(uint4*)e = v;
#pragma unroll
    for (int q = 0; q < 8; ++q) dst[(nc + q) * TP + k] = e[q];
  }
}

__global__ __launch_bounds__(128)
void k_gemm(const u16* __restrict__ A, const u16* __restrict__ Bm,
            const float* __restrict__ bias, float* __restrict__ Cf,
            u16* __restrict__ Cb, int R, int K, int N, int colBlks,
            long aBS, int aDiv, long bBS, long cBS, int flags, float alpha)
{
  __shared__ __align__(16) u16 sA[2][64 * TP];
  __shared__ __align__(16) u16 sB[2][64 * TP];

  const int batch   = blockIdx.y;
  const int rowBlk  = blockIdx.x / colBlks;
  const int colBlk  = blockIdx.x % colBlks;
  const int rowBase = rowBlk * 64, colBase = colBlk * 64;
  const u16* Ab = A + (long)(batch / aDiv) * aBS + (long)rowBase * K;
  const u16* Bb = Bm + (long)batch * bBS + colBase;
  const long cOff = (long)batch * cBS;

  const int tid = threadIdx.x, lane = tid & 31, wave = tid >> 5;

  v8f zero = {0.f,0.f,0.f,0.f,0.f,0.f,0.f,0.f};
  v8f acc[4] = {zero, zero, zero, zero};

  // prologue: fill slab 0
  stage_a(sA[0], Ab, K, tid);
  stage_b(sB[0], Bb, N, tid);
#ifdef ASYNC_LDS
  WAIT_ASYNC();
#endif
  __syncthreads();

  for (int k0 = 0; k0 < K; k0 += 64) {
    const int cur = (k0 >> 6) & 1, nxt = cur ^ 1;
    const bool more = (k0 + 64) < K;
    if (more) {
      stage_a(sA[nxt], Ab + (k0 + 64), K, tid);      // async fill of next slab
      __builtin_prefetch(Bb + (long)(k0 + 64) * N + (tid & 63), 0, 0);
    }

    // compute on current slab: 2 k-steps of 32, 4 column tiles each
#pragma unroll
    for (int ks = 0; ks < 64; ks += 32) {
      // A fragment (ISA 16-bit A 16x32 layout): elems 0-7 = K[kb..kb+7],
      // elems 8-15 = K[kb+16..kb+23], kb = ks + 8*(lane>>4), row m = lane&15.
      const int mloc = wave * 16 + (lane & 15);
      const int kb = ks + (lane >> 4) * 8;
      uint4 a0 = *(const uint4*)&sA[cur][mloc * TP + kb];
      uint4 a1 = *(const uint4*)&sA[cur][mloc * TP + kb + 16];
      v8u au = {a0.x, a0.y, a0.z, a0.w, a1.x, a1.y, a1.z, a1.w};
      v16bf aF = __builtin_bit_cast(v16bf, au);

      const int kh = ks + (lane >> 4) * 8;
#pragma unroll
      for (int tc = 0; tc < 4; ++tc) {
        int n = tc * 16 + (lane & 15);
        uint4 b0 = *(const uint4*)&sB[cur][n * TP + kh];
        uint4 b1 = *(const uint4*)&sB[cur][n * TP + kh + 16];
        v8u bu = {b0.x, b0.y, b0.z, b0.w, b1.x, b1.y, b1.z, b1.w};
        v16bf bF = __builtin_bit_cast(v16bf, bu);
        acc[tc] = __builtin_amdgcn_wmma_f32_16x16x32_bf16(
            false, aF, false, bF, (short)0, acc[tc], false, false);
      }
    }

    if (more) stage_b(sB[nxt], Bb + (long)(k0 + 64) * N, N, tid);
#ifdef ASYNC_LDS
    if (more) WAIT_ASYNC();
#endif
    __syncthreads();
  }

  // epilogue: C layout f32 16x16 => lane n=lane&15, m = v + 8*(lane>>4)
  const int mrow = rowBase + wave * 16 + 8 * (lane >> 4);
#pragma unroll
  for (int tc = 0; tc < 4; ++tc) {
    int n = colBase + tc * 16 + (lane & 15);
    float bv = (flags & GEMM_BIAS) ? bias[n] : 0.0f;
#pragma unroll
    for (int v = 0; v < 8; ++v) {
      long idx = cOff + (long)(mrow + v) * N + n;
      float val = alpha * acc[tc][v] + bv;
      if (flags & GEMM_ACC) val += Cf[idx];
      if (Cf) Cf[idx] = val;
      if (Cb) Cb[idx] = f2bf(val);
    }
  }
}

// ------------------------------ small kernels ------------------------------
__global__ void k_convert(const float* __restrict__ s, u16* __restrict__ d, long n) {
  long i = (long)blockIdx.x * blockDim.x + threadIdx.x;
  if (i < n) d[i] = f2bf(s[i]);
}
// dst[k*N+n] = src[n*ld + k + off]   (fp32 -> bf16 transposed weight prep)
__global__ void k_prep_wT(const float* __restrict__ src, u16* __restrict__ dst,
                          int K, int N, int ld, int off) {
  int i = blockIdx.x * blockDim.x + threadIdx.x;
  if (i >= K * N) return;
  int n = i % N, k = i / N;
  dst[i] = f2bf(src[(long)n * ld + k + off]);
}
// dst[b][C][R] = src[b][R][C]   (X[B,T,N] -> P_last bf16 [B,N,T])
__global__ void k_btranspose(const float* __restrict__ s, u16* __restrict__ d,
                             int batch, int R, int C) {
  long i = (long)blockIdx.x * blockDim.x + threadIdx.x;
  long tot = (long)batch * R * C;
  if (i >= tot) return;
  int c = (int)(i % C); long t = i / C; int r = (int)(t % R); int b = (int)(t / R);
  d[((long)b * C + c) * R + r] = f2bf(s[i]);
}
__global__ void k_softmax(const float* __restrict__ S, float* __restrict__ H,
                          int rows, int cols) {   // cols<=128, %32==0
  int row = (int)(((long)blockIdx.x * blockDim.x + threadIdx.x) >> 5);
  int lane = threadIdx.x & 31;
  if (row >= rows) return;
  const float* s = S + (long)row * cols;
  int ni = cols >> 5;
  float v[4], mx = -3.0e38f;
  for (int i = 0; i < ni; ++i) { v[i] = s[lane + 32 * i]; mx = fmaxf(mx, v[i]); }
  for (int o = 16; o; o >>= 1) mx = fmaxf(mx, __shfl_xor(mx, o, 32));
  float sum = 0.f;
  for (int i = 0; i < ni; ++i) { v[i] = __expf(v[i] - mx); sum += v[i]; }
  for (int o = 16; o; o >>= 1) sum += __shfl_xor(sum, o, 32);
  float inv = 1.0f / sum;
  float* h = H + (long)row * cols;
  for (int i = 0; i < ni; ++i) h[lane + 32 * i] = v[i] * inv;
}
__global__ void k_rowsum(const float* __restrict__ M, float* __restrict__ o,
                         int rows, int cols) {
  int r = blockIdx.x * blockDim.x + threadIdx.x;
  if (r >= rows) return;
  float s = 0.f;
  for (int c = 0; c < cols; ++c) s += M[(long)r * cols + c];
  o[r] = s;
}
__global__ void k_colsum_b(const float* __restrict__ M, float* __restrict__ o,
                           int batch, int rows, int cols) {
  int i = blockIdx.x * blockDim.x + threadIdx.x;
  if (i >= batch * cols) return;
  int c = i % cols, b = i / cols;
  const float* m = M + (long)b * rows * cols;
  float s = 0.f;
  for (int r = 0; r < rows; ++r) s += m[(long)r * cols + c];
  o[i] = s;
}
// Lb[b,r,c] = H * invD[b*rows+r] ; Rt[b,c,r] = H * invE[b*cols+c]  (both bf16)
__global__ void k_scale_pair(const float* __restrict__ H, const float* __restrict__ Ds,
                             const float* __restrict__ Es, u16* __restrict__ Lb,
                             u16* __restrict__ Rt, int batch, int rows, int cols) {
  long i = (long)blockIdx.x * blockDim.x + threadIdx.x;
  long tot = (long)batch * rows * cols;
  if (i >= tot) return;
  int c = (int)(i % cols); long t = i / cols; int r = (int)(t % rows); int b = (int)(t / rows);
  float h = H[i];
  float d = Ds[(long)b * rows + r], e = Es[(long)b * cols + c];
  float invd = d > 0.f ? 1.0f / (d + 1e-8f) : 0.0f;
  float inve = e > 0.f ? 1.0f / (e + 1e-8f) : 0.0f;
  Lb[i] = f2bf(h * invd);
  Rt[((long)b * cols + c) * rows + r] = f2bf(h * inve);
}
__global__ void k_start(const float* __restrict__ X, const float* __restrict__ sW,
                        const float* __restrict__ sb, float* __restrict__ xf,
                        u16* __restrict__ xb, float* __restrict__ sk, long n) {
  long i = (long)blockIdx.x * blockDim.x + threadIdx.x;
  if (i >= n) return;
  int d = (int)(i & 63);
  float v = X[i >> 6] * sW[d] + sb[d];
  xf[i] = v; xb[i] = f2bf(v); sk[i] = v;
}
__global__ void k_fg(const float* __restrict__ F, const float* __restrict__ G,
                     float* __restrict__ xf, u16* __restrict__ xb, long n) {
  long i = (long)blockIdx.x * blockDim.x + threadIdx.x;
  if (i >= n) return;
  float v = tanhf(F[i]) * sigm(G[i]);
  xf[i] = v; xb[i] = f2bf(v);
}
__global__ void k_add(float* __restrict__ d, const float* __restrict__ s, long n) {
  long i = (long)blockIdx.x * blockDim.x + threadIdx.x;
  if (i < n) d[i] += s[i];
}
__global__ void k_vblend(const float* __restrict__ vp, const float* __restrict__ xd,
                         const float* __restrict__ xg, u16* __restrict__ ob, long n) {
  long i = (long)blockIdx.x * blockDim.x + threadIdx.x;
  if (i >= n) return;
  float s = sigm(vp[i]);
  ob[i] = f2bf(s * xd[i] + (1.0f - s) * xg[i]);
}
__global__ void k_relu_bf16(const float* __restrict__ s, u16* __restrict__ d, long n) {
  long i = (long)blockIdx.x * blockDim.x + threadIdx.x;
  if (i < n) d[i] = f2bf(fmaxf(s[i], 0.0f));
}
__global__ void k_layernorm(const float* __restrict__ Xi, const float* __restrict__ g,
                            const float* __restrict__ b, float* __restrict__ Xf,
                            u16* __restrict__ Xb, int rows) {   // D == 64
  int row = (int)(((long)blockIdx.x * blockDim.x + threadIdx.x) >> 5);
  int lane = threadIdx.x & 31;
  if (row >= rows) return;
  const float* x = Xi + (long)row * 64;
  float a = x[lane], c = x[lane + 32];
  float s = a + c;
  for (int o = 16; o; o >>= 1) s += __shfl_xor(s, o, 32);
  float mean = s * (1.0f / 64.0f);
  float da = a - mean, dc = c - mean;
  float v = da * da + dc * dc;
  for (int o = 16; o; o >>= 1) v += __shfl_xor(v, o, 32);
  float inv = rsqrtf(v * (1.0f / 64.0f) + 1e-5f);
  float ra = da * inv * g[lane] + b[lane];
  float rc = dc * inv * g[lane + 32] + b[lane + 32];
  long off = (long)row * 64;
  Xf[off + lane] = ra;      Xf[off + lane + 32] = rc;
  Xb[off + lane] = f2bf(ra); Xb[off + lane + 32] = f2bf(rc);
}

// ------------------------------- launcher ----------------------------------
#define EW(kern, n, ...) \
  kern<<<dim3((unsigned)((((long)(n)) + 255) / 256)), dim3(256), 0, stream>>>(__VA_ARGS__)

extern "C" void kernel_launch(void* const* d_in, const int* in_sizes, int n_in,
                              void* d_out, int out_size, void* d_ws, size_t ws_size,
                              hipStream_t stream) {
  (void)in_sizes; (void)n_in; (void)out_size; (void)ws_size;
  const int Bn = 4, T = 64, Nn = 256, E = 128, D = 64;
  const long S = (long)Bn * T * Nn * D;       // 4,194,304
  const int RR = (int)(S / D);                // 65,536 rows

  const float *X = (const float*)d_in[0], *Amat = (const float*)d_in[1],
              *he = (const float*)d_in[2], *Gl = (const float*)d_in[3],
              *Wq = (const float*)d_in[4], *Wk = (const float*)d_in[5],
              *node_W = (const float*)d_in[6], *node_b = (const float*)d_in[7],
              *start_W = (const float*)d_in[8], *start_b = (const float*)d_in[9],
              *filt_W = (const float*)d_in[10], *filt_b = (const float*)d_in[11],
              *gate_W = (const float*)d_in[12], *gate_b = (const float*)d_in[13],
              *skip_W = (const float*)d_in[14], *skip_b = (const float*)d_in[15],
              *convW = (const float*)d_in[16], *convWg = (const float*)d_in[17],
              *Wfuse = (const float*)d_in[18], *Wgfuse = (const float*)d_in[19],
              *out_W = (const float*)d_in[20], *out_b = (const float*)d_in[21],
              *gamma = (const float*)d_in[22], *beta = (const float*)d_in[23],
              *end_W = (const float*)d_in[24], *end_b = (const float*)d_in[25];

  char* p = (char*)d_ws;
  auto aF = [&](long n) { float* r = (float*)p; p += ((n * 4 + 255) & ~255L); return r; };
  auto aH = [&](long n) { u16*   r = (u16*)p;   p += ((n * 2 + 255) & ~255L); return r; };

  // big fp32
  float *xf = aF(S), *skipf = aF(S), *t0 = aF(S), *t1 = aF(S), *xdf = aF(S), *xgf = aF(S);
  // big bf16
  u16 *xb = aH(S), *xdb = aH(S), *xgb = aH(S), *pb = aH(S), *o0b = aH(S), *o1b = aH(S);
  // constructor
  u16 *Pb = aH((long)Bn * Nn * T), *he_b = aH(E * D), *Eemb_b = aH((long)Bn * Nn * D),
      *Q_b = aH((long)Bn * Nn * D), *KmatT = aH(D * E), *Lb = aH((long)Bn * Nn * E),
      *Rt = aH((long)Bn * E * Nn), *Anorm_b = aH((long)Bn * Nn * Nn),
      *Ab_ = aH((long)Nn * Nn), *Glb = aH((long)Nn * E), *Lg = aH((long)Nn * E),
      *Rgt = aH((long)E * Nn), *Ageo_b = aH((long)Nn * Nn);
  float *Kmat_f = aF(E * D), *Scores = aF((long)Bn * Nn * E), *Hd = aF((long)Bn * Nn * E),
        *Dd = aF(Bn * Nn), *Bd = aF(Bn * E), *Hg = aF((long)Nn * E), *Dg = aF(Nn), *Bg = aF(E);
  // weights (bf16, [K,N] layout for GEMM)
  u16 *node_Wt = aH(T * D), *WqT = aH(D * D), *WkT = aH(D * D),
      *filtT = aH(3 * D * D), *gateT = aH(3 * D * D), *skipT = aH(3 * D * D),
      *convWb = aH(2 * D * D), *convWgb = aH(2 * D * D),
      *Wf1t = aH(D * D), *Wf2t = aH(D * D), *Wg1t = aH(D * D), *Wg2t = aH(D * D),
      *oWt0 = aH(D * D), *oWt1 = aH(D * D), *endT = aH(D * D);

  auto gemm = [&](const u16* A, const u16* Bm, const float* bias, float* Cf, u16* Cb,
                  int R, int K, int Ncol, int batches, long aBS, int aDiv,
                  long bBS, long cBS, int flags, float alpha) {
    dim3 g((unsigned)((R / 64) * (Ncol / 64)), (unsigned)batches);
    k_gemm<<<g, dim3(128), 0, stream>>>(A, Bm, bias, Cf, Cb, R, K, Ncol, Ncol / 64,
                                        aBS, aDiv, bBS, cBS, flags, alpha);
  };

  // ---- weight / input prep (fp32 -> bf16) ----
  EW(k_convert, E * D, he, he_b, (long)E * D);
  EW(k_convert, Nn * Nn, Amat, Ab_, (long)Nn * Nn);
  EW(k_convert, Nn * E, Gl, Glb, (long)Nn * E);
  for (int j = 0; j < 2; ++j) {
    EW(k_convert, D * D, convW + j * D * D, convWb + j * D * D, (long)D * D);
    EW(k_convert, D * D, convWg + j * D * D, convWgb + j * D * D, (long)D * D);
  }
  EW(k_prep_wT, T * D, node_W, node_Wt, T, D, T, 0);
  EW(k_prep_wT, D * D, Wq, WqT, D, D, D, 0);
  EW(k_prep_wT, D * D, Wk, WkT, D, D, D, 0);
  for (int i = 0; i < 3; ++i) {
    EW(k_prep_wT, D * D, filt_W + i * D * D, filtT + i * D * D, D, D, D, 0);
    EW(k_prep_wT, D * D, gate_W + i * D * D, gateT + i * D * D, D, D, D, 0);
    EW(k_prep_wT, D * D, skip_W + i * D * D, skipT + i * D * D, D, D, D, 0);
  }
  EW(k_prep_wT, D * D, Wfuse, Wf1t, D, D, 2 * D, 0);
  EW(k_prep_wT, D * D, Wfuse, Wf2t, D, D, 2 * D, D);
  EW(k_prep_wT, D * D, Wgfuse, Wg1t, D, D, 2 * D, 0);
  EW(k_prep_wT, D * D, Wgfuse, Wg2t, D, D, 2 * D, D);
  EW(k_prep_wT, D * D, out_W, oWt0, D, D, 2 * D, 0);
  EW(k_prep_wT, D * D, out_W, oWt1, D, D, 2 * D, D);
  EW(k_prep_wT, D * D, end_W, endT, D, D, D, 0);
  EW(k_btranspose, (long)Bn * T * Nn, X, Pb, Bn, T, Nn);

  // ---- hypergraph constructor ----
  gemm(he_b, WkT, nullptr, Kmat_f, nullptr, E, D, D, 1, 0, 1, 0, 0, 0, 1.f);
  EW(k_prep_wT, D * E, Kmat_f, KmatT, D, E, D, 0);
  gemm(Pb, node_Wt, node_b, nullptr, Eemb_b, Bn * Nn, T, D, 1, 0, 1, 0, 0, GEMM_BIAS, 1.f);
  gemm(Eemb_b, WqT, nullptr, nullptr, Q_b, Bn * Nn, D, D, 1, 0, 1, 0, 0, 0, 1.f);
  gemm(Q_b, KmatT, nullptr, Scores, nullptr, Bn * Nn, D, E, 1, 0, 1, 0, 0, 0, 0.125f);
  { int rows = Bn * Nn;
    k_softmax<<<(rows * 32 + 255) / 256, 256, 0, stream>>>(Scores, Hd, rows, E); }
  EW(k_rowsum, Bn * Nn, Hd, Dd, Bn * Nn, E);
  EW(k_colsum_b, Bn * E, Hd, Bd, Bn, Nn, E);
  EW(k_scale_pair, (long)Bn * Nn * E, Hd, Dd, Bd, Lb, Rt, Bn, Nn, E);
  gemm(Lb, Rt, nullptr, nullptr, Anorm_b, Nn, E, Nn, Bn,
       (long)Nn * E, 1, (long)E * Nn, (long)Nn * Nn, 0, 1.f);
  gemm(Ab_, Glb, nullptr, Hg, nullptr, Nn, Nn, E, 1, 0, 1, 0, 0, 0, 1.f);
  EW(k_rowsum, Nn, Hg, Dg, Nn, E);
  EW(k_colsum_b, E, Hg, Bg, 1, Nn, E);
  EW(k_scale_pair, (long)Nn * E, Hg, Dg, Bg, Lg, Rgt, 1, Nn, E);
  gemm(Lg, Rgt, nullptr, nullptr, Ageo_b, Nn, E, Nn, 1, 0, 1, 0, 0, 0, 1.f);

  // ---- start linear ----
  EW(k_start, S, X, start_W, start_b, xf, xb, skipf, S);

  // ---- temporal + hypergraph layers ----
  for (int L = 0; L < 3; ++L) {
    gemm(xb, filtT + L * D * D, filt_b + L * D, t0, nullptr, RR, D, D, 1, 0, 1, 0, 0, GEMM_BIAS, 1.f);
    gemm(xb, gateT + L * D * D, gate_b + L * D, t1, nullptr, RR, D, D, 1, 0, 1, 0, 0, GEMM_BIAS, 1.f);
    EW(k_fg, S, t0, t1, xf, xb, S);
    gemm(xb, skipT + L * D * D, skip_b + L * D, t0, nullptr, RR, D, D, 1, 0, 1, 0, 0, GEMM_BIAS, 1.f);
    EW(k_add, S, skipf, t0, S);

    const u16* xdc = xb; const u16* xgc = xb;
    for (int j = 0; j < 2; ++j) {
      // xd <- (A_norm[b] @ xd) @ convW[j]
      gemm(Anorm_b, xdc, nullptr, nullptr, pb, Nn, Nn, D, Bn * T,
           (long)Nn * Nn, T, (long)Nn * D, (long)Nn * D, 0, 1.f);
      gemm(pb, convWb + j * D * D, nullptr, xdf, xdb, RR, D, D, 1, 0, 1, 0, 0, 0, 1.f);
      // xg <- (A_geo @ xg) @ convWg[j]
      gemm(Ageo_b, xgc, nullptr, nullptr, pb, Nn, Nn, D, Bn * T,
           0, 1, (long)Nn * D, (long)Nn * D, 0, 1.f);
      gemm(pb, convWgb + j * D * D, nullptr, xgf, xgb, RR, D, D, 1, 0, 1, 0, 0, 0, 1.f);
      // Vpre = xd@Wf1 + x@Wf2 + xg@Wg1 + x@Wg2
      gemm(xdb, Wf1t, nullptr, t0, nullptr, RR, D, D, 1, 0, 1, 0, 0, 0, 1.f);
      gemm(xb,  Wf2t, nullptr, t0, nullptr, RR, D, D, 1, 0, 1, 0, 0, GEMM_ACC, 1.f);
      gemm(xgb, Wg1t, nullptr, t0, nullptr, RR, D, D, 1, 0, 1, 0, 0, GEMM_ACC, 1.f);
      gemm(xb,  Wg2t, nullptr, t0, nullptr, RR, D, D, 1, 0, 1, 0, 0, GEMM_ACC, 1.f);
      EW(k_vblend, S, t0, xdf, xgf, (j == 0 ? o0b : o1b), S);
      xdc = xdb; xgc = xgb;
    }
    gemm(o0b, oWt0, out_b, t1, nullptr, RR, D, D, 1, 0, 1, 0, 0, GEMM_BIAS, 1.f);
    gemm(o1b, oWt1, nullptr, t1, nullptr, RR, D, D, 1, 0, 1, 0, 0, GEMM_ACC, 1.f);
    k_layernorm<<<(RR * 32 + 255) / 256, 256, 0, stream>>>(t1, gamma, beta, xf, xb, RR);
  }

  // ---- output: relu(skip) @ end_W.T + end_b ----
  EW(k_relu_bf16, S, skipf, xb, S);
  gemm(xb, endT, end_b, (float*)d_out, nullptr, RR, D, D, 1, 0, 1, 0, 0, GEMM_BIAS, 1.f);
}
